// LocalAttention_26207890440894
// MI455X (gfx1250) — matrix operand order repaired and checked by reading, not compiled, not run on hardware
//
#include <hip/hip_runtime.h>
#include <hip/hip_bf16.h>
#include <cstdint>

typedef __attribute__((ext_vector_type(16))) _Float16 v16h;
typedef __attribute__((ext_vector_type(8)))  _Float16 v8h;
typedef __attribute__((ext_vector_type(8)))  float    v8f;

union Frag { v16h v; v8h h[2]; };

// ---- WMMA wrapper: D = A(16x32 f16) * B(32x16 f16) + C(16x16 f32) ----
__device__ __forceinline__ v8f wmma_f16(v16h a, v16h b, v8f c) {
  return __builtin_amdgcn_wmma_f32_16x16x32_f16(
      /*neg_a=*/false, a, /*neg_b=*/false, b,
      /*c_mod=*/(short)0, c, /*reuse_a=*/false, /*reuse_b=*/false);
}

// ---- A-fragment loader (16x32, row-major source, rows = M, cols = K) ----
// ISA layout: lanes 0-15 row M=lane, K chunks {k0..k0+7, k0+16..k0+23}
//             lanes 16-31 row M=lane-16, K chunks {k0+8..15, k0+24..31}
__device__ __forceinline__ v16h load_afrag(const _Float16* __restrict__ p,
                                           int row0, int k0, int ld) {
  const int lane = threadIdx.x & 31;
  const _Float16* base =
      p + (size_t)(row0 + (lane & 15)) * ld + k0 + ((lane >> 4) << 3);
  Frag f;
  f.h[0] = *(const v8h*)(base);
  f.h[1] = *(const v8h*)(base + 16);
  return f.v;
}

// ---- B-fragment loader (32x16 KxN; source row-major with rows = N, cols = K;
//      i.e. B[k][n] = src[n][k]) ----
// ISA layout: lanes 0-15 col N=lane, K chunks {k0..k0+7, k0+16..k0+23};
//             lanes 16-31 col N=lane-16, K chunks {k0+8..15, k0+24..31}
__device__ __forceinline__ v16h load_bfrag(const _Float16* __restrict__ p,
                                           int col0, int k0, int ld) {
  const int lane = threadIdx.x & 31;
  const _Float16* base =
      p + (size_t)(col0 + (lane & 15)) * ld + k0 + ((lane >> 4) << 3);
  Frag f;
  f.h[0] = *(const v8h*)(base);
  f.h[1] = *(const v8h*)(base + 16);
  return f.v;
}

// ---- async 16B global->LDS copy (gfx1250 GLOBAL_LOAD_ASYNC_TO_LDS_B128) ----
__device__ __forceinline__ void async_copy_b128(uint32_t lds_byte_addr,
                                                uint64_t glob_byte_addr) {
  asm volatile("global_load_async_to_lds_b128 %0, %1, off"
               :: "v"(lds_byte_addr), "v"(glob_byte_addr) : "memory");
}
__device__ __forceinline__ void async_wait0() {
  asm volatile("s_wait_asynccnt 0" ::: "memory");
}

// ============================ kernels ============================

__global__ void cvt_f32_to_f16_kernel(const float* __restrict__ src,
                                      _Float16* __restrict__ dst, int n) {
  int i = blockIdx.x * blockDim.x + threadIdx.x;
  if (i < n) dst[i] = (_Float16)src[i];
}

// Fused QKV projection: y = x @ W^T + b  (M=4096, N=1024, K=1024)
// Register-blocked: each wave computes a 64x32 output tile (4x2 WMMA tiles).
// which==0 -> Q (scaled by 1/8) stored [bh][t][64]
// which==1 -> K stored [bh][t][64]
// which==2 -> V stored transposed [bh][64][t]
__global__ __launch_bounds__(256)
void qkv_proj_kernel(const _Float16* __restrict__ x16,
                     const _Float16* __restrict__ Wq16,
                     const _Float16* __restrict__ Wk16,
                     const _Float16* __restrict__ Wv16,
                     const float* __restrict__ bq,
                     const float* __restrict__ bk,
                     const float* __restrict__ bv,
                     _Float16* __restrict__ Q16,
                     _Float16* __restrict__ K16,
                     _Float16* __restrict__ Vt16) {
  const int wave = threadIdx.x >> 5;
  const int lane = threadIdx.x & 31;
  const int tile = blockIdx.x * 8 + wave;
  const int tilesPer = (4096 / 64) * (1024 / 32);  // 2048
  const int which = tile / tilesPer;
  const int t2 = tile % tilesPer;
  const int m0 = (t2 / 32) * 64;
  const int n0 = (t2 % 32) * 32;

  const _Float16* W = (which == 0) ? Wq16 : (which == 1) ? Wk16 : Wv16;
  const float* bias = (which == 0) ? bq : (which == 1) ? bk : bv;

  v8f acc[4][2] = {};
#pragma unroll 2
  for (int k0 = 0; k0 < 1024; k0 += 32) {
    v16h a0 = load_afrag(x16, m0,      k0, 1024);
    v16h a1 = load_afrag(x16, m0 + 16, k0, 1024);
    v16h a2 = load_afrag(x16, m0 + 32, k0, 1024);
    v16h a3 = load_afrag(x16, m0 + 48, k0, 1024);
    v16h b0 = load_bfrag(W, n0,      k0, 1024);
    v16h b1 = load_bfrag(W, n0 + 16, k0, 1024);
    acc[0][0] = wmma_f16(a0, b0, acc[0][0]);
    acc[0][1] = wmma_f16(a0, b1, acc[0][1]);
    acc[1][0] = wmma_f16(a1, b0, acc[1][0]);
    acc[1][1] = wmma_f16(a1, b1, acc[1][1]);
    acc[2][0] = wmma_f16(a2, b0, acc[2][0]);
    acc[2][1] = wmma_f16(a2, b1, acc[2][1]);
    acc[3][0] = wmma_f16(a3, b0, acc[3][0]);
    acc[3][1] = wmma_f16(a3, b1, acc[3][1]);
  }

  const float scale = (which == 0) ? 0.125f : 1.0f;  // fold 1/sqrt(64) into Q
#pragma unroll
  for (int j = 0; j < 2; ++j) {
    const int n = n0 + 16 * j + (lane & 15);
    const float bn = bias[n];
    const int h = n >> 6, d = n & 63;
#pragma unroll
    for (int i = 0; i < 4; ++i) {
      const int mbase = m0 + 16 * i + ((lane >> 4) << 3);
#pragma unroll
      for (int r = 0; r < 8; ++r) {
        const int m = mbase + r;
        const int bb = m >> 11, t = m & 2047;
        const _Float16 hv = (_Float16)((acc[i][j][r] + bn) * scale);
        if (which == 2) {
          Vt16[((size_t)(bb * 16 + h) * 64 + d) * 2048 + t] = hv;
        } else {
          _Float16* dst = (which == 0) ? Q16 : K16;
          dst[((size_t)(bb * 16 + h) * 2048 + t) * 64 + d] = hv;
        }
      }
    }
  }
}

// Flash attention with anti-band mask keep = |i-j| > 16.
// Block = 8 waves sharing one (b,h); waves hold distinct 16-query tiles but
// walk the same 64-key steps, so K/V tiles are staged once per block into LDS
// with async global->LDS copies (ASYNCcnt), then consumed via ds_load frags.
__global__ __launch_bounds__(256)
void attn_kernel(const _Float16* __restrict__ Q16,
                 const _Float16* __restrict__ K16,
                 const _Float16* __restrict__ Vt16,
                 _Float16* __restrict__ AO16) {
  __shared__ __align__(16) _Float16 Ktile[64 * 64];   // [key][d]   8 KB
  __shared__ __align__(16) _Float16 Vtile[64 * 64];   // [d][key]   8 KB
  __shared__ __align__(16) _Float16 pshare[8][16 * 64];  // P tiles 16 KB
  const int tid = threadIdx.x;
  const int wave = tid >> 5;
  const int lane = tid & 31;
  const int bh = blockIdx.x;                    // 0..31
  const int q0 = (blockIdx.y * 8 + wave) * 16;  // query tile base in [0,2048)

  const _Float16* Qb = Q16 + (size_t)bh * 2048 * 64;
  const _Float16* Kb = K16 + (size_t)bh * 2048 * 64;
  const _Float16* Vb = Vt16 + (size_t)bh * 64 * 2048;

  // Q fragments (16x64 = two 16x32 A-frags), pre-scaled by 1/sqrt(D)
  const v16h qa0 = load_afrag(Qb, q0, 0, 64);
  const v16h qa1 = load_afrag(Qb, q0, 32, 64);

  v8f o0 = {}, o1 = {}, o2 = {}, o3 = {};  // O: 16x64
  const float NEGINF = -__builtin_inff();
  float mrow[8], lrow[8];
  for (int r = 0; r < 8; ++r) { mrow[r] = NEGINF; lrow[r] = 0.0f; }

  const int myq = q0 + ((lane >> 4) << 3);  // this lane's row-stat base
  const uint32_t kldsbase = (uint32_t)(uintptr_t)(&Ktile[0]);
  const uint32_t vldsbase = (uint32_t)(uintptr_t)(&Vtile[0]);

  for (int k0 = 0; k0 < 2048; k0 += 64) {
    // ---- async stage K tile (contiguous 8KB) and V tile (64 rows x 128B) ---
    {
      const uint64_t kg = (uint64_t)(uintptr_t)(Kb + (size_t)k0 * 64);
      const uint64_t vg = (uint64_t)(uintptr_t)(Vb + (size_t)k0);
#pragma unroll
      for (int i = 0; i < 2; ++i) {
        const uint32_t c = (uint32_t)tid + 256u * i;   // 512 x 16B chunks
        async_copy_b128(kldsbase + c * 16u, kg + (uint64_t)c * 16u);
        async_copy_b128(vldsbase + c * 16u,
                        vg + (uint64_t)(c >> 3) * 4096u + (uint64_t)(c & 7) * 16u);
      }
      async_wait0();       // my chunks landed in LDS
      __syncthreads();     // everyone's chunks landed
    }
    // ---- scores: S = Qs @ K^T  (four 16x16 C tiles covering 64 keys) ----
    v8f s[4];
#pragma unroll
    for (int t = 0; t < 4; ++t) {
      v8f st = {};
      st = wmma_f16(qa0, load_bfrag(Ktile, 16 * t, 0, 64), st);
      st = wmma_f16(qa1, load_bfrag(Ktile, 16 * t, 32, 64), st);
      s[t] = st;
    }
    // ---- anti-band mask: drop entries with |qi - kj| <= 16 (uniform test) --
    if (k0 <= q0 + 31 && k0 >= q0 - 79) {
      const int kj = k0 + (lane & 15);
#pragma unroll
      for (int t = 0; t < 4; ++t) {
#pragma unroll
        for (int r = 0; r < 8; ++r) {
          int dd = (myq + r) - (kj + 16 * t);
          if (dd < 0) dd = -dd;
          if (dd <= 16) s[t][r] = NEGINF;
        }
      }
    }
    // ---- online softmax (row stats per VGPR index, reduced over 16 lanes) --
    float pscale[8];
#pragma unroll
    for (int r = 0; r < 8; ++r) {
      float t = fmaxf(fmaxf(s[0][r], s[1][r]), fmaxf(s[2][r], s[3][r]));
      t = fmaxf(t, __shfl_xor(t, 1, 32));
      t = fmaxf(t, __shfl_xor(t, 2, 32));
      t = fmaxf(t, __shfl_xor(t, 4, 32));
      t = fmaxf(t, __shfl_xor(t, 8, 32));
      const float mnew = fmaxf(mrow[r], t);
      const float mt = fmaxf(mnew, -1e30f);   // NaN guard for all-masked rows
      const float sc = __expf(mrow[r] - mt);  // -inf -> 0
      mrow[r] = mnew;
      float ps = 0.0f;
#pragma unroll
      for (int tt = 0; tt < 4; ++tt) {
        const float p = __expf(s[tt][r] - mt);
        s[tt][r] = p;
        ps += p;
      }
      ps += __shfl_xor(ps, 1, 32);
      ps += __shfl_xor(ps, 2, 32);
      ps += __shfl_xor(ps, 4, 32);
      ps += __shfl_xor(ps, 8, 32);
      lrow[r] = lrow[r] * sc + ps;
      pscale[r] = sc;
    }
#pragma unroll
    for (int r = 0; r < 8; ++r) {
      const float sc = pscale[r];
      o0[r] *= sc; o1[r] *= sc; o2[r] *= sc; o3[r] *= sc;
    }
    // ---- P (C-layout) -> LDS -> A-fragment layout (16x64) ----
    _Float16* ps_ = pshare[wave];
    const int prow = (lane >> 4) << 3;
    const int pcol = lane & 15;
#pragma unroll
    for (int r = 0; r < 8; ++r) {
#pragma unroll
      for (int t = 0; t < 4; ++t)
        ps_[(prow + r) * 64 + pcol + 16 * t] = (_Float16)s[t][r];
    }
    __builtin_amdgcn_wave_barrier();
    asm volatile("s_wait_dscnt 0" ::: "memory");  // wave-internal LDS visibility
    const v16h pa0 = load_afrag(ps_, 0, 0, 64);
    const v16h pa1 = load_afrag(ps_, 0, 32, 64);
    __builtin_amdgcn_wave_barrier();
    // ---- O += P(16x64) @ V(64x64)  (Vtile rows = d, cols = key) ----
    o0 = wmma_f16(pa0, load_bfrag(Vtile, 0,  0, 64), o0);
    o1 = wmma_f16(pa0, load_bfrag(Vtile, 16, 0, 64), o1);
    o2 = wmma_f16(pa0, load_bfrag(Vtile, 32, 0, 64), o2);
    o3 = wmma_f16(pa0, load_bfrag(Vtile, 48, 0, 64), o3);
    o0 = wmma_f16(pa1, load_bfrag(Vtile, 0,  32, 64), o0);
    o1 = wmma_f16(pa1, load_bfrag(Vtile, 16, 32, 64), o1);
    o2 = wmma_f16(pa1, load_bfrag(Vtile, 32, 32, 64), o2);
    o3 = wmma_f16(pa1, load_bfrag(Vtile, 48, 32, 64), o3);
    __syncthreads();     // K/V tiles fully consumed before next stage
  }

  // ---- normalize and store to AO16 [4096][1024], col = h*64 + d ----
  const int h = bh & 15, bb = bh >> 4;
  const int col0 = h * 64;
  const int mg0 = bb * 2048 + q0 + ((lane >> 4) << 3);
  const int cn = lane & 15;
#pragma unroll
  for (int r = 0; r < 8; ++r) {
    const float inv = 1.0f / lrow[r];  // every row keeps >= T-33 keys
    const size_t ro = (size_t)(mg0 + r) * 1024;
    AO16[ro + col0 + cn]      = (_Float16)(o0[r] * inv);
    AO16[ro + col0 + 16 + cn] = (_Float16)(o1[r] * inv);
    AO16[ro + col0 + 32 + cn] = (_Float16)(o2[r] * inv);
    AO16[ro + col0 + 48 + cn] = (_Float16)(o3[r] * inv);
  }
}

// Output projection: out = AO @ Wo^T + bo (f32 result)
// Register-blocked: each wave computes a 64x32 output tile (4x2 WMMA tiles).
__global__ __launch_bounds__(256)
void out_proj_kernel(const _Float16* __restrict__ AO16,
                     const _Float16* __restrict__ Wo16,
                     const float* __restrict__ bo,
                     float* __restrict__ out) {
  const int wave = threadIdx.x >> 5;
  const int lane = threadIdx.x & 31;
  const int tile = blockIdx.x * 8 + wave;
  const int m0 = (tile / 32) * 64;
  const int n0 = (tile % 32) * 32;

  v8f acc[4][2] = {};
#pragma unroll 2
  for (int k0 = 0; k0 < 1024; k0 += 32) {
    v16h a0 = load_afrag(AO16, m0,      k0, 1024);
    v16h a1 = load_afrag(AO16, m0 + 16, k0, 1024);
    v16h a2 = load_afrag(AO16, m0 + 32, k0, 1024);
    v16h a3 = load_afrag(AO16, m0 + 48, k0, 1024);
    v16h b0 = load_bfrag(Wo16, n0,      k0, 1024);
    v16h b1 = load_bfrag(Wo16, n0 + 16, k0, 1024);
    acc[0][0] = wmma_f16(a0, b0, acc[0][0]);
    acc[0][1] = wmma_f16(a0, b1, acc[0][1]);
    acc[1][0] = wmma_f16(a1, b0, acc[1][0]);
    acc[1][1] = wmma_f16(a1, b1, acc[1][1]);
    acc[2][0] = wmma_f16(a2, b0, acc[2][0]);
    acc[2][1] = wmma_f16(a2, b1, acc[2][1]);
    acc[3][0] = wmma_f16(a3, b0, acc[3][0]);
    acc[3][1] = wmma_f16(a3, b1, acc[3][1]);
  }
#pragma unroll
  for (int j = 0; j < 2; ++j) {
    const int n = n0 + 16 * j + (lane & 15);
    const float bn = bo[n];
#pragma unroll
    for (int i = 0; i < 4; ++i) {
      const int mb = m0 + 16 * i + ((lane >> 4) << 3);
#pragma unroll
      for (int r = 0; r < 8; ++r)
        out[(size_t)(mb + r) * 1024 + n] = acc[i][j][r] + bn;
    }
  }
}

// ============================ host launch ============================

extern "C" void kernel_launch(void* const* d_in, const int* in_sizes, int n_in,
                              void* d_out, int out_size, void* d_ws, size_t ws_size,
                              hipStream_t stream) {
  const float* x  = (const float*)d_in[0];
  const float* Wq = (const float*)d_in[1];
  const float* bq = (const float*)d_in[2];
  const float* Wk = (const float*)d_in[3];
  const float* bk = (const float*)d_in[4];
  const float* Wv = (const float*)d_in[5];
  const float* bv = (const float*)d_in[6];
  const float* Wo = (const float*)d_in[7];
  const float* bo = (const float*)d_in[8];
  float* out = (float*)d_out;

  const size_t NX = 4096ull * 1024;     // x / AO elements
  const size_t NW = 1024ull * 1024;     // weight elements
  const size_t NH = 32ull * 2048 * 64;  // per Q/K/V head-tensor elements (== NX)

  _Float16* ws   = (_Float16*)d_ws;
  _Float16* x16  = ws;
  _Float16* Wq16 = x16 + NX;
  _Float16* Wk16 = Wq16 + NW;
  _Float16* Wv16 = Wk16 + NW;
  _Float16* Wo16 = Wv16 + NW;
  _Float16* Q16  = Wo16 + NW;
  _Float16* K16  = Q16 + NH;
  _Float16* Vt16 = K16 + NH;
  _Float16* AO16 = Vt16 + NH;

  // f32 -> f16 conversions
  cvt_f32_to_f16_kernel<<<(int)((NX + 255) / 256), 256, 0, stream>>>(x, x16, (int)NX);
  cvt_f32_to_f16_kernel<<<(int)((NW + 255) / 256), 256, 0, stream>>>(Wq, Wq16, (int)NW);
  cvt_f32_to_f16_kernel<<<(int)((NW + 255) / 256), 256, 0, stream>>>(Wk, Wk16, (int)NW);
  cvt_f32_to_f16_kernel<<<(int)((NW + 255) / 256), 256, 0, stream>>>(Wv, Wv16, (int)NW);
  cvt_f32_to_f16_kernel<<<(int)((NW + 255) / 256), 256, 0, stream>>>(Wo, Wo16, (int)NW);

  // QKV projection: 3 * (64 m-tiles * 32 n-tiles) waves, 8 waves/block
  qkv_proj_kernel<<<3 * 2048 / 8, 256, 0, stream>>>(
      x16, Wq16, Wk16, Wv16, bq, bk, bv, Q16, K16, Vt16);

  // Flash attention: 32 bh * 128 q-tiles, 8 waves/block
  attn_kernel<<<dim3(32, 16), 256, 0, stream>>>(Q16, K16, Vt16, AO16);

  // Output projection: 64*32 tiles, 8 waves/block
  out_proj_kernel<<<2048 / 8, 256, 0, stream>>>(AO16, Wo16, bo, out);
}
